// NetVLAD_4415226380607
// MI455X (gfx1250) — matrix operand order repaired and checked by reading, not compiled, not run on hardware
//
#include <hip/hip_runtime.h>
#include <hip/hip_bf16.h>

// ---------------------------------------------------------------------------
// NetVLAD forward, fused for MI455X (gfx1250, wave32, WMMA bf16)
// ---------------------------------------------------------------------------

typedef __bf16 bf16_t;
typedef __attribute__((ext_vector_type(16))) __bf16 v16bf;
typedef __attribute__((ext_vector_type(8)))  __bf16 v8bf;
typedef __attribute__((ext_vector_type(4)))  __bf16 v4bf;
typedef __attribute__((ext_vector_type(8)))  float  v8f;
typedef __attribute__((ext_vector_type(4)))  float  v4f;

#define N_IMG   32
#define C_DIM   256
#define S_DIM   4096
#define K_CL    64
#define OUT_DIM 1024
#define KC_DIM  (K_CL * C_DIM)   // 16384

#define ST          64           // s-tile width per WMMA pass
#define NCHUNK      16           // S split into 16 chunks of 256
#define S_PER_CHUNK 256
#define TS_PER_CHUNK 4           // 4 tiles of 64 per chunk

#define T1 512                   // kernel-1 block size: 16 waves

// LDS pitches (elements), padded to break 64-bank conflicts
#define WPITCH 264   // conv_w rows (256 + 8)
#define XCP    72    // x c-major rows (64 + 8)
#define XTP    264   // x s-major rows (256 + 8)
#define LGP    66    // logits f32 rows (64 + 2)
#define AP     72    // assignment rows (64 + 8)

// LDS byte offsets for kernel 1 (dynamic LDS)
#define OFF_W   0
#define SZ_W    (K_CL * WPITCH * 2)          // 33792
#define OFF_XC  (OFF_W + SZ_W)               // 33792
#define SZ_XC   (C_DIM * XCP * 2)            // 36864
#define OFF_UNI (OFF_XC + SZ_XC)             // 70656
#define SZ_XT   (ST * XTP * 2)               // 33792 (union member 1)
#define OFF_LG  OFF_UNI                      // union member 2: logits f32
#define SZ_LG   (K_CL * LGP * 4)             // 16896
#define OFF_A   (OFF_UNI + SZ_LG)            // 87552: assignment bf16
#define SZ_A    (K_CL * AP * 2)              // 9216
#define LDS1_BYTES (OFF_UNI + SZ_XT)         // 104448

#define EPSN 1e-12f

// ---- fragment loaders (contiguous k = 16*half + i in both A and B, 16-bit operands) -----

// A operand: lane (l&15) = row; lane half selects contiguous 16-wide K chunk
__device__ __forceinline__ v16bf load_afrag(const bf16_t* base, int row,
                                            int pitch, int k0, int half) {
  const bf16_t* p = base + row * pitch + k0;
  const v8bf lo = *reinterpret_cast<const v8bf*>(p + 16 * half);
  const v8bf hi = *reinterpret_cast<const v8bf*>(p + 16 * half + 8);
  v16bf r;
#pragma unroll
  for (int i = 0; i < 8; ++i) { r[i] = lo[i]; r[i + 8] = hi[i]; }
  return r;
}

__device__ __forceinline__ v16bf load_afrag_ptr(const bf16_t* rowp, int k0, int half) {
  const v8bf lo = *reinterpret_cast<const v8bf*>(rowp + k0 + 16 * half);
  const v8bf hi = *reinterpret_cast<const v8bf*>(rowp + k0 + 16 * half + 8);
  v16bf r;
#pragma unroll
  for (int i = 0; i < 8; ++i) { r[i] = lo[i]; r[i + 8] = hi[i]; }
  return r;
}

// B operand: lane (l&15) = column; lane half selects contiguous 16-wide K chunk
__device__ __forceinline__ v16bf load_bfrag(const bf16_t* base, int col,
                                            int pitch, int k0, int half) {
  const bf16_t* p = base + col * pitch + k0 + 16 * half;
  const v8bf lo = *reinterpret_cast<const v8bf*>(p);
  const v8bf hi = *reinterpret_cast<const v8bf*>(p + 8);
  v16bf r;
#pragma unroll
  for (int i = 0; i < 8; ++i) { r[i] = lo[i]; r[i + 8] = hi[i]; }
  return r;
}

// B operand sourced from f32 global memory via 4x global_load_b128 + cvt
__device__ __forceinline__ v16bf load_bfrag_f32(const float* rowp, int k0, int half) {
  const v4f* p4 = reinterpret_cast<const v4f*>(rowp + k0 + 16 * half);
  const v4f q0 = p4[0], q1 = p4[1], q2 = p4[2], q3 = p4[3];
  v16bf r;
#pragma unroll
  for (int i = 0; i < 4; ++i) {
    r[i]      = (bf16_t)q0[i];
    r[i + 4]  = (bf16_t)q1[i];
    r[i + 8]  = (bf16_t)q2[i];
    r[i + 12] = (bf16_t)q3[i];
  }
  return r;
}

__device__ __forceinline__ v8f wmma_bf16(v16bf a, v16bf b, v8f c) {
  // D = A(16x32) * B(32x16) + C, f32 accumulate
  return __builtin_amdgcn_wmma_f32_16x16x32_bf16(false, a, false, b,
                                                 (short)0, c, false, false);
}

// ---------------------------------------------------------------------------
// Kernel 1: fused logits -> softmax -> vlad partial aggregation
// grid: 512 blocks = 32 images x 16 S-chunks; 512 threads (16 waves)
// ---------------------------------------------------------------------------
__global__ void __launch_bounds__(T1, 1)
nv_k1_fused(const float* __restrict__ x,
            const float* __restrict__ conv_w,
            float* __restrict__ vladPart,
            float* __restrict__ asumPart) {
  extern __shared__ char smem[];
  bf16_t* ldsW  = (bf16_t*)(smem + OFF_W);
  bf16_t* ldsXc = (bf16_t*)(smem + OFF_XC);
  bf16_t* ldsXT = (bf16_t*)(smem + OFF_UNI);
  float*  lg    = (float*)(smem + OFF_LG);
  bf16_t* ldsA  = (bf16_t*)(smem + OFF_A);

  const int tid    = threadIdx.x;
  const int lane   = tid & 31;
  const int wv     = tid >> 5;        // 0..15
  const int lanelo = lane & 15;
  const int half   = lane >> 4;

  const int n     = blockIdx.x >> 4;
  const int chunk = blockIdx.x & 15;

  // stage conv_w as bf16 in LDS: float4 loads, packed 4x-bf16 stores
#pragma unroll
  for (int i = tid; i < (K_CL * C_DIM) / 4; i += T1) {   // 8 iterations
    const v4f q = reinterpret_cast<const v4f*>(conv_w)[i];
    const int e = i * 4;
    const int k = e >> 8, c = e & 255;
    v4bf b;
#pragma unroll
    for (int u = 0; u < 4; ++u) b[u] = (bf16_t)q[u];
    *reinterpret_cast<v4bf*>(&ldsW[k * WPITCH + c]) = b;
  }

  // persistent vlad accumulators: 4 tiles of 16x16 per wave -> 64x256 total
  v8f vacc[4];
  v8f zero = {};
#pragma unroll
  for (int j = 0; j < 4; ++j) vacc[j] = zero;
  float asumLoc = 0.f;

  const float* xn = x + (size_t)n * C_DIM * S_DIM;

  // logits tile for this wave: 16 waves <-> 16 tiles of the 64x64 output
  const int m0l = (wv >> 2) * 16;
  const int scl = (wv & 3) * 16;
  const int m0v = (wv >> 2) * 16;     // vlad row block: (tt>>4)*16, tt=wv*4+j2

  for (int ts = 0; ts < TS_PER_CHUNK; ++ts) {
    __syncthreads();   // prior-iteration LDS consumers done
    const int s0 = chunk * S_PER_CHUNK + ts * ST;

    // x tile: float4 global loads (8 independent 16B loads in flight/thread),
    // packed store to c-major tile + 4 scalar stores to transposed tile
#pragma unroll
    for (int i = tid; i < (C_DIM * ST) / 4; i += T1) {   // 8 iterations
      const int e = i * 4;
      const int c = e >> 6, s = e & 63;
      const v4f q = *reinterpret_cast<const v4f*>(xn + (size_t)c * S_DIM + s0 + s);
      v4bf b;
#pragma unroll
      for (int u = 0; u < 4; ++u) b[u] = (bf16_t)q[u];
      *reinterpret_cast<v4bf*>(&ldsXc[c * XCP + s]) = b;   // B of vlad GEMM
#pragma unroll
      for (int u = 0; u < 4; ++u)
        ldsXT[(s + u) * XTP + c] = b[u];                   // B of logits GEMM
    }
    __syncthreads();

    // logits = conv_w @ x_tile : one 16x16 tile per wave, K=256 -> 8 steps
    v8f lacc = zero;
#pragma unroll
    for (int cc = 0; cc < C_DIM; cc += 64) {
      const v16bf a0 = load_afrag(ldsW,  m0l + lanelo, WPITCH, cc,      half);
      const v16bf b0 = load_bfrag(ldsXT, scl + lanelo, XTP,    cc,      half);
      const v16bf a1 = load_afrag(ldsW,  m0l + lanelo, WPITCH, cc + 32, half);
      const v16bf b1 = load_bfrag(ldsXT, scl + lanelo, XTP,    cc + 32, half);
      lacc = wmma_bf16(a0, b0, lacc);
      lacc = wmma_bf16(a1, b1, lacc);
    }
    __syncthreads();   // all XT reads complete before union region is reused

    // spill logits (f32) into union region
#pragma unroll
    for (int j = 0; j < 8; ++j)
      lg[(m0l + j + 8 * half) * LGP + scl + lanelo] = lacc[j];
    __syncthreads();

    // softmax over K=64 clusters, one column per thread (f32)
    if (tid < ST) {
      const int s = tid;
      float mx = -3.402823466e38f;
      for (int k = 0; k < K_CL; ++k) mx = fmaxf(mx, lg[k * LGP + s]);
      float sum = 0.f;
      for (int k = 0; k < K_CL; ++k) {
        const float e = __expf(lg[k * LGP + s] - mx);
        lg[k * LGP + s] = e;
        sum += e;
      }
      const float inv = 1.f / sum;
      for (int k = 0; k < K_CL; ++k)
        ldsA[k * AP + s] = (bf16_t)(lg[k * LGP + s] * inv);
    }
    __syncthreads();

    // deterministic per-k assignment sums (no float atomics)
    if (tid < K_CL) {
      for (int s = 0; s < ST; ++s) asumLoc += (float)ldsA[tid * AP + s];
    }

    // vlad += a_tile(64xST) @ x_tile^T(STx256): 64 tiles, 4 per wave, K=64
    const v16bf av0 = load_afrag(ldsA, m0v + lanelo, AP, 0,  half);
    const v16bf av1 = load_afrag(ldsA, m0v + lanelo, AP, 32, half);
#pragma unroll
    for (int j2 = 0; j2 < 4; ++j2) {
      const int c0 = (((wv & 3) * 4 + j2) & 15) * 16;   // (tt&15)*16
      const v16bf bv0 = load_bfrag(ldsXc, c0 + lanelo, XCP, 0,  half);
      const v16bf bv1 = load_bfrag(ldsXc, c0 + lanelo, XCP, 32, half);
      vacc[j2] = wmma_bf16(av0, bv0, vacc[j2]);
      vacc[j2] = wmma_bf16(av1, bv1, vacc[j2]);
    }
  }

  // write per-chunk vlad partials (f32) -> workspace
  float* vp = vladPart + ((size_t)(chunk * N_IMG + n)) * K_CL * C_DIM;
#pragma unroll
  for (int j2 = 0; j2 < 4; ++j2) {
    const int c0 = (((wv & 3) * 4 + j2) & 15) * 16;
#pragma unroll
    for (int j = 0; j < 8; ++j)
      vp[(m0v + j + 8 * half) * C_DIM + c0 + lanelo] = vacc[j2][j];
  }
  if (tid < K_CL)
    asumPart[(chunk * N_IMG + n) * K_CL + tid] = asumLoc;
}

// ---------------------------------------------------------------------------
// Kernel 2: chunk reduction + centroid subtract + intra-cluster L2 norm
// grid: 2048 blocks (n,k), 256 threads (c)
// ---------------------------------------------------------------------------
__global__ void nv_k2_intra(const float* __restrict__ vladPart,
                            const float* __restrict__ asumPart,
                            const float* __restrict__ centroids,
                            bf16_t* __restrict__ vladN,
                            float* __restrict__ rowSS) {
  const int nk = blockIdx.x;
  const int n = nk >> 6, k = nk & 63;
  const int c = threadIdx.x;

  float r = 0.f;
  for (int ch = 0; ch < NCHUNK; ++ch)
    r += vladPart[(((size_t)ch * N_IMG + n) * K_CL + k) * C_DIM + c];

  __shared__ float s_asum;
  __shared__ float red[256];
  if (c == 0) {
    float a = 0.f;
    for (int ch = 0; ch < NCHUNK; ++ch)
      a += asumPart[(ch * N_IMG + n) * K_CL + k];
    s_asum = a;
  }
  __syncthreads();

  r -= s_asum * centroids[k * C_DIM + c];

  red[c] = r * r;
  __syncthreads();
  for (int off = 128; off > 0; off >>= 1) {
    if (c < off) red[c] += red[c + off];
    __syncthreads();
  }
  const float norm2 = red[0];
  const float inv = 1.f / fmaxf(sqrtf(norm2), EPSN);

  vladN[(size_t)n * KC_DIM + k * C_DIM + c] = (bf16_t)(r * inv);
  if (c == 0) rowSS[nk] = norm2 * inv * inv;   // ||normalized row||^2
}

// ---------------------------------------------------------------------------
// Kernel 2b: per-image global L2 scale (folded into the linear layer)
// ---------------------------------------------------------------------------
__global__ void nv_k2b_scale(const float* __restrict__ rowSS,
                             float* __restrict__ invg) {
  const int n = blockIdx.x, t = threadIdx.x;   // 64 threads
  __shared__ float red[64];
  red[t] = rowSS[n * K_CL + t];
  __syncthreads();
  for (int off = 32; off > 0; off >>= 1) {
    if (t < off) red[t] += red[t + off];
    __syncthreads();
  }
  if (t == 0) invg[n] = 1.f / fmaxf(sqrtf(red[0]), EPSN);
}

// ---------------------------------------------------------------------------
// Kernel 3: out_pre = (invg[n] * vladN) @ lin_w^T + lin_b
// [32 x 16384] x [16384 x 1024], bf16 WMMA, split-K over 8 waves per block
// ---------------------------------------------------------------------------
__global__ void __launch_bounds__(256)
nv_k3_linear(const bf16_t* __restrict__ vladN,
             const float* __restrict__ lin_w,
             const float* __restrict__ lin_b,
             const float* __restrict__ invg,
             float* __restrict__ out_pre) {
  __shared__ float part[8 * 2 * 256];   // 16 KB: [wave][mtile][lane*8+j]

  const int tid    = threadIdx.x;
  const int lane   = tid & 31;
  const int wv     = tid >> 5;
  const int lanelo = lane & 15;
  const int half   = lane >> 4;
  const int o0     = blockIdx.x * 16;

  v8f acc0 = {}, acc1 = {};
  const float*  w0  = lin_w + (size_t)(o0 + lanelo) * KC_DIM;
  const bf16_t* a0p = vladN + (size_t)lanelo * KC_DIM;        // rows n=0..15
  const bf16_t* a1p = vladN + (size_t)(16 + lanelo) * KC_DIM; // rows n=16..31
  const int kbase = wv * (KC_DIM / 8);                        // 2048 per wave

  for (int kk = 0; kk < KC_DIM / 8; kk += 32) {
    const int kcur = kbase + kk;
    const v16bf bf  = load_bfrag_f32(w0, kcur, half);   // lin_w read once
    const v16bf af0 = load_afrag_ptr(a0p, kcur, half);
    const v16bf af1 = load_afrag_ptr(a1p, kcur, half);
    acc0 = wmma_bf16(af0, bf, acc0);
    acc1 = wmma_bf16(af1, bf, acc1);
  }

#pragma unroll
  for (int j = 0; j < 8; ++j) {
    part[(wv * 2 + 0) * 256 + lane * 8 + j] = acc0[j];
    part[(wv * 2 + 1) * 256 + lane * 8 + j] = acc1[j];
  }
  __syncthreads();

  // deterministic cross-wave split-K reduction + epilogue
  for (int f = tid; f < 512; f += 256) {
    const int t  = f >> 8;
    const int r  = f & 255;
    const int ln = r >> 3;
    const int j  = r & 7;
    float s = 0.f;
#pragma unroll
    for (int w = 0; w < 8; ++w) s += part[(w * 2 + t) * 256 + r];
    const int nrow = 16 * t + j + 8 * (ln >> 4);
    const int o    = o0 + (ln & 15);
    out_pre[nrow * OUT_DIM + o] = s * invg[nrow] + lin_b[o];
  }
}

// ---------------------------------------------------------------------------
// Kernel 4: final per-image L2 normalize of the 1024-dim output
// ---------------------------------------------------------------------------
__global__ void nv_k4_outnorm(const float* __restrict__ out_pre,
                              float* __restrict__ out) {
  const int n = blockIdx.x, t = threadIdx.x;   // 256 threads
  __shared__ float red[256];
  float ss = 0.f;
  for (int o = t; o < OUT_DIM; o += 256) {
    const float v = out_pre[n * OUT_DIM + o];
    ss += v * v;
  }
  red[t] = ss;
  __syncthreads();
  for (int off = 128; off > 0; off >>= 1) {
    if (t < off) red[t] += red[t + off];
    __syncthreads();
  }
  const float inv = 1.f / fmaxf(sqrtf(red[0]), EPSN);
  for (int o = t; o < OUT_DIM; o += 256)
    out[n * OUT_DIM + o] = out_pre[n * OUT_DIM + o] * inv;
}

// ---------------------------------------------------------------------------
extern "C" void kernel_launch(void* const* d_in, const int* in_sizes, int n_in,
                              void* d_out, int out_size, void* d_ws, size_t ws_size,
                              hipStream_t stream) {
  const float* x         = (const float*)d_in[0];
  const float* conv_w    = (const float*)d_in[1];
  const float* centroids = (const float*)d_in[2];
  const float* lin_w     = (const float*)d_in[3];
  const float* lin_b     = (const float*)d_in[4];
  float* out = (float*)d_out;

  // workspace carve (every region fully overwritten each call; no pre-zero)
  char* ws = (char*)d_ws;
  float*  vladPart = (float*) (ws);                      // 16*32*64*256*4 = 32 MiB
  float*  asumPart = (float*) (ws + 33554432);           // 16*32*64*4     = 128 KiB
  bf16_t* vladN    = (bf16_t*)(ws + 33685504);           // 32*16384*2     = 1 MiB
  float*  rowSS    = (float*) (ws + 34734080);           // 32*64*4
  float*  invg     = (float*) (ws + 34742272);           // 32*4
  float*  out_pre  = (float*) (ws + 34742400);           // 32*1024*4

  nv_k1_fused<<<dim3(N_IMG * NCHUNK), dim3(T1), LDS1_BYTES, stream>>>(
      x, conv_w, vladPart, asumPart);
  nv_k2_intra<<<dim3(N_IMG * K_CL), dim3(256), 0, stream>>>(
      vladPart, asumPart, centroids, vladN, rowSS);
  nv_k2b_scale<<<dim3(N_IMG), dim3(64), 0, stream>>>(rowSS, invg);
  nv_k3_linear<<<dim3(OUT_DIM / 16), dim3(256), 0, stream>>>(
      vladN, lin_w, lin_b, invg, out_pre);
  nv_k4_outnorm<<<dim3(N_IMG), dim3(256), 0, stream>>>(out_pre, out);
}
